// RNNLayer_79353815761263
// MI455X (gfx1250) — matrix operand order repaired and checked by reading, not compiled, not run on hardware
//
#include <hip/hip_runtime.h>
#include <cstdint>

// LSTM: B=64, T=512, I=H=1024.
// Phase 1 (parallel): pregates[t,b,:] = x[b,t,:] @ Wih^T, bf16 WMMA GEMM, chunked over T (TC=64).
// Phase 2 (serial):   ONE persistent kernel per chunk runs all 64 steps; 32 WGs, each with its
//                     Whh slice resident in LDS (CDNA5 320KB/WGP), inter-WG step barrier via
//                     agent-scope fences + global atomic counter.
#define BB     64
#define TT     512
#define II     1024
#define HH     1024
#define G4     4096
#define TC     64
#define NWG    32     // workgroups in persistent recurrent kernel (32 h-cols each)
#define LROW   20     // LDS pitch (dwords) of a 32-bf16 staging row: 16B-aligned + conflict-free
#define BPITCH 516    // resident Whh row pitch in dwords (512 data + 4 pad)
#define GPITCH 132    // gates LDS row pitch in floats (128 + 4 pad)

typedef __bf16 bf16;
typedef __bf16 v16bf __attribute__((ext_vector_type(16)));
typedef float  v8f   __attribute__((ext_vector_type(8)));

struct U32x8 { uint4 lo, hi; };

// A fragment (16x32 bf16 MxK): lane m = lane&15; two 16B reads at dword offsets 4*lh, 8+4*lh.
static __device__ __forceinline__ v16bf frag_a(const unsigned* row, int lh) {
  U32x8 u;
  u.lo = *reinterpret_cast<const uint4*>(row + 4 * lh);
  u.hi = *reinterpret_cast<const uint4*>(row + 8 + 4 * lh);
  return __builtin_bit_cast(v16bf, u);
}
// B fragment (32x16 bf16 KxN): lane n = lane&15; two 16B reads at dword offsets 4*lh, 8+4*lh.
static __device__ __forceinline__ v16bf frag_b(const unsigned* row, int lh) {
  U32x8 u;
  u.lo = *reinterpret_cast<const uint4*>(row + 4 * lh);
  u.hi = *reinterpret_cast<const uint4*>(row + 8 + 4 * lh);
  return __builtin_bit_cast(v16bf, u);
}
static __device__ __forceinline__ v8f zero8() {
  v8f z;
#pragma unroll
  for (int e = 0; e < 8; ++e) z[e] = 0.0f;
  return z;
}
static __device__ __forceinline__ float sigf(float x) { return 1.0f / (1.0f + __expf(-x)); }

// ---------------- fp32 -> bf16, 4 elements/thread ----------------
__global__ __launch_bounds__(256) void cvt_bf16x4_kernel(const float4* __restrict__ in,
                                                         uint2* __restrict__ outp, int n4) {
  const int i = blockIdx.x * 256 + threadIdx.x;
  if (i < n4) {
    const float4 f = in[i];
    union { bf16 h[4]; uint2 u; } c;
    c.h[0] = (bf16)f.x; c.h[1] = (bf16)f.y; c.h[2] = (bf16)f.z; c.h[3] = (bf16)f.w;
    outp[i] = c.u;
  }
}

// ---------------- zero h0 ping-pong, c0, step-barrier counters ----------------
__global__ __launch_bounds__(256) void init_state_kernel(bf16* __restrict__ hA,
                                                         bf16* __restrict__ hB,
                                                         float* __restrict__ c,
                                                         unsigned* __restrict__ syncc) {
  const int i = blockIdx.x * 256 + threadIdx.x;  // grid covers B*H
  hA[i] = (bf16)0.0f;
  hB[i] = (bf16)0.0f;
  c[i]  = 0.0f;
  if (i < TT) syncc[i] = 0u;
}

// ---------------- Phase 1: pregates chunk GEMM (M=4096, N=4096, K=1024) ----------------
// 128x128 tile per WG (8 waves: wave = m-block, 8 n-tiles each), double-buffered LDS staging.
// launch_bounds(...,1): relax occupancy target so the staged tiles live in VGPRs (no spills).
__global__ __launch_bounds__(256, 1) void pregate_gemm_kernel(const bf16* __restrict__ xbf,
                                                              const bf16* __restrict__ wihb,
                                                              float* __restrict__ pre, int t0) {
  __shared__ unsigned Au[2][128 * LROW];
  __shared__ unsigned Bu[2][128 * LROW];
  const int tid  = threadIdx.x;
  const int wave = tid >> 5, lane = tid & 31;
  const int lm = lane & 15, lh = lane >> 4;
  const int m0 = blockIdx.y * 128;
  const int n0 = blockIdx.x * 128;

  // Per-thread global source pointers for its two 16B chunks of A and B (row r, chunk q).
  const int r0 = tid >> 2, q0 = tid & 3;           // chunks 0..255
  const int r1 = (tid + 256) >> 2, q1 = tid & 3;   // chunks 256..511
  const int rowA0 = m0 + r0, rowA1 = m0 + r1;      // chunk row = tt*64 + bb
  const bf16* gA0 = xbf + ((size_t)(rowA0 & 63) * TT + (t0 + (rowA0 >> 6))) * II + q0 * 8;
  const bf16* gA1 = xbf + ((size_t)(rowA1 & 63) * TT + (t0 + (rowA1 >> 6))) * II + q1 * 8;
  const bf16* gB0 = wihb + (size_t)(n0 + r0) * II + q0 * 8;
  const bf16* gB1 = wihb + (size_t)(n0 + r1) * II + q1 * 8;

  v8f acc[8];
#pragma unroll
  for (int i = 0; i < 8; ++i) acc[i] = zero8();

  uint4 ar0 = *reinterpret_cast<const uint4*>(gA0);
  uint4 ar1 = *reinterpret_cast<const uint4*>(gA1);
  uint4 br0 = *reinterpret_cast<const uint4*>(gB0);
  uint4 br1 = *reinterpret_cast<const uint4*>(gB1);

  int p = 0;
  for (int k0 = 0; k0 < II; k0 += 32) {
    *reinterpret_cast<uint4*>(&Au[p][r0 * LROW + q0 * 4]) = ar0;
    *reinterpret_cast<uint4*>(&Au[p][r1 * LROW + q1 * 4]) = ar1;
    *reinterpret_cast<uint4*>(&Bu[p][r0 * LROW + q0 * 4]) = br0;
    *reinterpret_cast<uint4*>(&Bu[p][r1 * LROW + q1 * 4]) = br1;
    const int kn = (k0 + 32 < II) ? k0 + 32 : k0;  // branch-free; last reload is dead
    ar0 = *reinterpret_cast<const uint4*>(gA0 + kn);
    ar1 = *reinterpret_cast<const uint4*>(gA1 + kn);
    br0 = *reinterpret_cast<const uint4*>(gB0 + kn);
    br1 = *reinterpret_cast<const uint4*>(gB1 + kn);
    __syncthreads();
    const v16bf a = frag_a(&Au[p][(wave * 16 + lm) * LROW], lh);
#pragma unroll
    for (int nt = 0; nt < 8; ++nt) {
      const v16bf b = frag_b(&Bu[p][(nt * 16 + lm) * LROW], lh);
      acc[nt] = __builtin_amdgcn_wmma_f32_16x16x32_bf16(false, a, false, b,
                                                        (short)0, acc[nt], false, false);
    }
    p ^= 1;
  }
#pragma unroll
  for (int nt = 0; nt < 8; ++nt) {
    const int col = n0 + nt * 16 + lm;
#pragma unroll
    for (int r = 0; r < 8; ++r)
      pre[(size_t)(m0 + wave * 16 + r + 8 * lh) * G4 + col] = acc[nt][r];
  }
}

// ---------------- Phase 2: persistent recurrent kernel (one launch = TC steps) ----------------
// WG owns 32 h-columns (128 gate cols). Whh slice (128x1024 bf16, 258KB) LDS-resident.
// Per step: gates(64x128) = h(64x1024) @ Whh_slice^T via WMMA, + pregates + bias -> cell.
// Inter-WG step barrier: release fence -> atomic arrive -> spin -> acquire fence (global_inv).
__global__ __launch_bounds__(256, 1) void lstm_seq_kernel(const bf16* __restrict__ whhb,
                                                          const float* __restrict__ pre,
                                                          const float* __restrict__ bias,
                                                          float* __restrict__ cbuf,
                                                          bf16* __restrict__ hA,
                                                          bf16* __restrict__ hB,
                                                          float* __restrict__ out,
                                                          unsigned* __restrict__ syncc, int t0) {
  extern __shared__ unsigned dsm[];
  unsigned* Bw   = dsm;                                  // 128 * BPITCH  (258KB resident Whh)
  unsigned* Au   = dsm + 128 * BPITCH;                   // 2 * 64 * LROW (A staging, double buf)
  float*    gate = (float*)(dsm + 128 * BPITCH + 2 * 64 * LROW);  // 64 * GPITCH

  const int tid  = threadIdx.x;
  const int wave = tid >> 5, lane = tid & 31;
  const int lm = lane & 15, lh = lane >> 4;
  const int mb = wave >> 1, nh = wave & 1;   // 4 m-blocks x 2 n-halves, 4 n-tiles each
  const int nbase = blockIdx.x * 32;

  // Load resident Whh slice: rows n=0..127 -> Whh[gate*H + nbase + j], B-frag layout.
  for (int c = tid; c < 128 * 128; c += 256) {
    const int n = c >> 7, ck = c & 127;
    const int kb = ck >> 2, q = ck & 3;
    const int gt = n >> 5, j = n & 31;
    const uint4 v = *reinterpret_cast<const uint4*>(
        whhb + (size_t)(gt * HH + nbase + j) * HH + ck * 8);
    *reinterpret_cast<uint4*>(&Bw[n * BPITCH + kb * 16 + q * 4]) = v;
  }

  // Cell-state + bias held in registers across all TC steps.
  const int m  = tid >> 2;            // row (batch) 0..63  (also A-staging row)
  const int q  = tid & 3;             // A-staging chunk
  const int j0 = q * 8;               // 8 contiguous h-columns per thread
  float breg[4][8], creg[8];
#pragma unroll
  for (int g = 0; g < 4; ++g) {
    const float4 lo = *reinterpret_cast<const float4*>(bias + g * HH + nbase + j0);
    const float4 hi = *reinterpret_cast<const float4*>(bias + g * HH + nbase + j0 + 4);
    breg[g][0] = lo.x; breg[g][1] = lo.y; breg[g][2] = lo.z; breg[g][3] = lo.w;
    breg[g][4] = hi.x; breg[g][5] = hi.y; breg[g][6] = hi.z; breg[g][7] = hi.w;
  }
  {
    const float4 lo = *reinterpret_cast<const float4*>(cbuf + (size_t)m * HH + nbase + j0);
    const float4 hi = *reinterpret_cast<const float4*>(cbuf + (size_t)m * HH + nbase + j0 + 4);
    creg[0] = lo.x; creg[1] = lo.y; creg[2] = lo.z; creg[3] = lo.w;
    creg[4] = hi.x; creg[5] = hi.y; creg[6] = hi.z; creg[7] = hi.w;
  }
  __syncthreads();

  for (int tt = 0; tt < TC; ++tt) {
    const int t = t0 + tt;
    const bf16* hin  = (t & 1) ? hB : hA;
    bf16*       hout = (t & 1) ? hA : hB;

    v8f acc[4];
#pragma unroll
    for (int i = 0; i < 4; ++i) acc[i] = zero8();

    // K loop: stream h (64x1024) through double-buffered LDS; weights already resident.
    uint4 areg = *reinterpret_cast<const uint4*>(hin + (size_t)m * HH + j0);  // kb=0 chunk
    int p = 0;
    for (int kb = 0; kb < 32; ++kb) {
      *reinterpret_cast<uint4*>(&Au[p * 64 * LROW + m * LROW + q * 4]) = areg;
      const int kn = (kb + 1 < 32) ? kb + 1 : kb;
      areg = *reinterpret_cast<const uint4*>(hin + (size_t)m * HH + kn * 32 + j0);
      __syncthreads();
      const v16bf a = frag_a(&Au[p * 64 * LROW + (mb * 16 + lm) * LROW], lh);
#pragma unroll
      for (int nt = 0; nt < 4; ++nt) {
        const v16bf b = frag_b(&Bw[((nh * 4 + nt) * 16 + lm) * BPITCH + kb * 16], lh);
        acc[nt] = __builtin_amdgcn_wmma_f32_16x16x32_bf16(false, a, false, b,
                                                          (short)0, acc[nt], false, false);
      }
      p ^= 1;
    }

    // Accumulators -> LDS gate buffer [64][128]
#pragma unroll
    for (int nt = 0; nt < 4; ++nt) {
      const int c = (nh * 4 + nt) * 16 + lm;
#pragma unroll
      for (int rr = 0; rr < 8; ++rr)
        gate[(mb * 16 + rr + 8 * lh) * GPITCH + c] = acc[nt][rr];
    }
    __syncthreads();

    // LSTM cell: thread -> (row m, 8 contiguous cols), vectorized pregate loads.
    const float* pg = pre + ((size_t)tt * BB + m) * G4 + nbase + j0;
    float pr[4][8];
#pragma unroll
    for (int g = 0; g < 4; ++g) {
      const float4 lo = *reinterpret_cast<const float4*>(pg + g * HH);
      const float4 hi = *reinterpret_cast<const float4*>(pg + g * HH + 4);
      pr[g][0] = lo.x; pr[g][1] = lo.y; pr[g][2] = lo.z; pr[g][3] = lo.w;
      pr[g][4] = hi.x; pr[g][5] = hi.y; pr[g][6] = hi.z; pr[g][7] = hi.w;
    }
    if (tt + 1 < TC) __builtin_prefetch(pg + (size_t)BB * G4, 0, 1);  // next step's pregates

    float hv[8];
    union { bf16 h[8]; uint4 u; } hu;
#pragma unroll
    for (int jj = 0; jj < 8; ++jj) {
      const float ig = gate[m * GPITCH + 0 * 32 + j0 + jj] + pr[0][jj] + breg[0][jj];
      const float fg = gate[m * GPITCH + 1 * 32 + j0 + jj] + pr[1][jj] + breg[1][jj];
      const float gg = gate[m * GPITCH + 2 * 32 + j0 + jj] + pr[2][jj] + breg[2][jj];
      const float og = gate[m * GPITCH + 3 * 32 + j0 + jj] + pr[3][jj] + breg[3][jj];
      const float cn = sigf(fg) * creg[jj] + sigf(ig) * tanhf(gg);
      const float hn = sigf(og) * tanhf(cn);
      creg[jj] = cn;
      hv[jj] = hn;
      hu.h[jj] = (bf16)hn;
    }
    *reinterpret_cast<uint4*>(hout + (size_t)m * HH + nbase + j0) = hu.u;
    float* op = out + ((size_t)m * TT + t) * HH + nbase + j0;
    *reinterpret_cast<float4*>(op)     = make_float4(hv[0], hv[1], hv[2], hv[3]);
    *reinterpret_cast<float4*>(op + 4) = make_float4(hv[4], hv[5], hv[6], hv[7]);
    if (t == TT - 1) {  // final (h_T, c_T) appended after outputs
      float* ht = out + (size_t)BB * TT * HH + (size_t)m * HH + nbase + j0;
      float* ct = ht + (size_t)BB * HH;
      *reinterpret_cast<float4*>(ht)     = make_float4(hv[0], hv[1], hv[2], hv[3]);
      *reinterpret_cast<float4*>(ht + 4) = make_float4(hv[4], hv[5], hv[6], hv[7]);
      *reinterpret_cast<float4*>(ct)     = make_float4(creg[0], creg[1], creg[2], creg[3]);
      *reinterpret_cast<float4*>(ct + 4) = make_float4(creg[4], creg[5], creg[6], creg[7]);
    }

    // Inter-workgroup step barrier (not needed after the last step of the chunk).
    if (tt + 1 < TC) {
      __builtin_amdgcn_fence(__ATOMIC_RELEASE, "agent");   // drain stores to coherence point
      __syncthreads();
      if (tid == 0) {
        __hip_atomic_fetch_add(&syncc[t], 1u, __ATOMIC_RELAXED, __HIP_MEMORY_SCOPE_AGENT);
        while (__hip_atomic_load(&syncc[t], __ATOMIC_RELAXED, __HIP_MEMORY_SCOPE_AGENT) < NWG)
          __builtin_amdgcn_s_sleep(1);
      }
      __syncthreads();
      __builtin_amdgcn_fence(__ATOMIC_ACQUIRE, "agent");   // global_inv: no stale h in WGP$
    }
  }

  // Persist cell state for the next chunk's kernel.
  float* cp = cbuf + (size_t)m * HH + nbase + j0;
  *reinterpret_cast<float4*>(cp)     = make_float4(creg[0], creg[1], creg[2], creg[3]);
  *reinterpret_cast<float4*>(cp + 4) = make_float4(creg[4], creg[5], creg[6], creg[7]);
}

extern "C" void kernel_launch(void* const* d_in, const int* in_sizes, int n_in,
                              void* d_out, int out_size, void* d_ws, size_t ws_size,
                              hipStream_t stream) {
  (void)in_sizes; (void)n_in; (void)out_size; (void)ws_size;
  const float* x    = (const float*)d_in[0];
  const float* Wih  = (const float*)d_in[1];
  const float* Whh  = (const float*)d_in[2];
  const float* bias = (const float*)d_in[3];
  float* out = (float*)d_out;

  // Workspace carve-up (~152 MB)
  char* p = (char*)d_ws;
  bf16*  xbf  = (bf16*)p;  p += (size_t)BB * TT * II * sizeof(bf16);   // 67 MB
  bf16*  wihb = (bf16*)p;  p += (size_t)G4 * II * sizeof(bf16);        // 8.4 MB
  bf16*  whhb = (bf16*)p;  p += (size_t)G4 * HH * sizeof(bf16);        // 8.4 MB
  bf16*  hA   = (bf16*)p;  p += (size_t)BB * HH * sizeof(bf16);
  bf16*  hB   = (bf16*)p;  p += (size_t)BB * HH * sizeof(bf16);
  float* cbuf = (float*)p; p += (size_t)BB * HH * sizeof(float);
  unsigned* syncc = (unsigned*)p; p += (size_t)TT * sizeof(unsigned) + 1024;
  float* pre  = (float*)p; p += (size_t)TC * BB * G4 * sizeof(float);  // 67 MB

  cvt_bf16x4_kernel<<<(BB * TT * II / 4) / 256, 256, 0, stream>>>(
      (const float4*)x, (uint2*)xbf, BB * TT * II / 4);
  cvt_bf16x4_kernel<<<(G4 * II / 4) / 256, 256, 0, stream>>>(
      (const float4*)Wih, (uint2*)wihb, G4 * II / 4);
  cvt_bf16x4_kernel<<<(G4 * HH / 4) / 256, 256, 0, stream>>>(
      (const float4*)Whh, (uint2*)whhb, G4 * HH / 4);
  init_state_kernel<<<(BB * HH) / 256, 256, 0, stream>>>(hA, hB, cbuf, syncc);

  const size_t seq_lds = (size_t)(128 * BPITCH + 2 * 64 * LROW + 64 * GPITCH) * 4;  // ~301 KB
  for (int c = 0; c < TT / TC; ++c) {
    pregate_gemm_kernel<<<dim3(G4 / 128, (TC * BB) / 128), 256, 0, stream>>>(
        xbf, wihb, pre, c * TC);
    lstm_seq_kernel<<<NWG, 256, seq_lds, stream>>>(whhb, pre, bias, cbuf, hA, hB, out,
                                                   syncc, c * TC);
  }
}
